// SelfAttentionMasked_40716289966629
// MI455X (gfx1250) — matrix operand order repaired and checked by reading, not compiled, not run on hardware
//
#include <hip/hip_runtime.h>
#include <stdint.h>

// ---------------------------------------------------------------------------
// CDNA5 (gfx1250) masked multi-head self-attention, bf16 WMMA pipeline.
// b=4, n=2048, dim=1024, heads=16, dim_head=64.
// ---------------------------------------------------------------------------

typedef __bf16 bf16;
typedef float v8f  __attribute__((ext_vector_type(8)));
typedef bf16  v16bf __attribute__((ext_vector_type(16)));

union ABFrag { uint32_t u[8]; v16bf v; };

__device__ __forceinline__ v8f wmma_bf16(v16bf a, v16bf b, v8f c) {
  // D = A(16x32 bf16) * B(32x16 bf16) + C(16x16 f32)
  return __builtin_amdgcn_wmma_f32_16x16x32_bf16(
      /*neg_a=*/false, a, /*neg_b=*/false, b,
      /*c_mod=*/(short)0, c, /*reuse_a=*/false, /*reuse_b=*/false);
}

// A fragment (16x32, bf16) from row-major storage, element (m,k) at base[m*ld+k].
// ISA layout: lanes 0-15 row M=lane, VGPR p<4 -> K=2p,2p+1, p>=4 -> K=16+...,
// lanes 16-31 same rows with K offset +8 in each half.
__device__ __forceinline__ v16bf load_a_rowmajor(const bf16* base, int ld) {
  int lane = threadIdx.x & 31;
  int m  = lane & 15;
  int hi = lane >> 4;
  const uint32_t* p = (const uint32_t*)(base + (size_t)m * ld);
  ABFrag f;
#pragma unroll
  for (int i = 0; i < 8; ++i)
    f.u[i] = p[((i < 4) ? 0 : 8) + hi * 4 + (i & 3)];   // dword index = K/2
  return f.v;
}

// B fragment (32x16, bf16), element (k,n) stored K-contiguous at base[n*ld+k].
// ISA layout: lane holds column N=lane&15; VGPR p<4 -> K=2p,2p+1, p>=4 -> K=16+...,
// lanes 16-31 same with K offset +8 in each half.
__device__ __forceinline__ v16bf load_b_kmajor(const bf16* base, int ld) {
  int lane = threadIdx.x & 31;
  int n  = lane & 15;
  int hi = lane >> 4;
  const uint32_t* p = (const uint32_t*)(base + (size_t)n * ld);
  ABFrag f;
#pragma unroll
  for (int i = 0; i < 8; ++i) f.u[i] = p[((i < 4) ? 0 : 8) + hi * 4 + (i & 3)];
  return f.v;
}

// ---------------------------------------------------------------------------
// fp32 -> bf16 conversion kernels
// ---------------------------------------------------------------------------
__global__ void k_cvt(const float* __restrict__ x, bf16* __restrict__ y, int n) {
  int i = blockIdx.x * blockDim.x + threadIdx.x;
  if (i < n) y[i] = (bf16)x[i];
}

// w[k][n] (row-major KxN) -> wt[n][k] (K-contiguous per output column)
__global__ void k_cvt_t(const float* __restrict__ w, bf16* __restrict__ wt,
                        int K, int N) {
  int i = blockIdx.x * blockDim.x + threadIdx.x;
  if (i < K * N) {
    int k = i / N, n = i % N;
    wt[(size_t)n * K + k] = (bf16)w[i];
  }
}

// ---------------------------------------------------------------------------
// QKV GEMM: qkv[8192,3072] = xb[8192,1024] @ WqkvT^T.  Per-wave tile 32x64.
// Q (cols 0..1023)   -> qk[row][col]        pre-scaled by 1/sqrt(64)=0.125
// K (cols 1024..2047)-> qk[row][col]
// V (cols 2048..3071)-> vt[b][h][d][n]  (transposed, key-contiguous)
// ---------------------------------------------------------------------------
__global__ __launch_bounds__(256)
void k_gemm_qkv(const bf16* __restrict__ xb, const bf16* __restrict__ wT,
                bf16* __restrict__ qk, bf16* __restrict__ vt) {
  const int K = 1024, tilesN = 48;  // 3072/64
  int wid = blockIdx.x * (blockDim.x >> 5) + (threadIdx.x >> 5);
  int tm = wid / tilesN, tn = wid % tilesN;
  int m0 = tm * 32, n0 = tn * 64;
  int lane = threadIdx.x & 31, hi = lane >> 4, ln = lane & 15;
  v8f acc[2][4] = {};
  for (int k0 = 0; k0 < K; k0 += 32) {
    v16bf a0 = load_a_rowmajor(xb + (size_t)m0 * K + k0, K);
    v16bf a1 = load_a_rowmajor(xb + (size_t)(m0 + 16) * K + k0, K);
#pragma unroll
    for (int j = 0; j < 4; ++j) {
      v16bf b = load_b_kmajor(wT + (size_t)(n0 + 16 * j) * K + k0, K);
      acc[0][j] = wmma_bf16(a0, b, acc[0][j]);
      acc[1][j] = wmma_bf16(a1, b, acc[1][j]);
    }
  }
#pragma unroll
  for (int mi = 0; mi < 2; ++mi)
#pragma unroll
    for (int j = 0; j < 4; ++j) {
      int col = n0 + 16 * j + ln;
#pragma unroll
      for (int r = 0; r < 8; ++r) {
        int row = m0 + mi * 16 + r + 8 * hi;
        float val = acc[mi][j][r];
        if (col < 1024) {
          qk[(size_t)row * 2048 + col] = (bf16)(val * 0.125f);   // Q * scale
        } else if (col < 2048) {
          qk[(size_t)row * 2048 + col] = (bf16)val;              // K
        } else {
          int hd = col - 2048;                                    // h*64 + d
          int bb = row >> 11, t = row & 2047;
          vt[((size_t)(bb * 16 + (hd >> 6)) * 64 + (hd & 63)) * 2048 + t] =
              (bf16)val;                                          // V^T
        }
      }
    }
}

// ---------------------------------------------------------------------------
// Flash attention: one wave = one (b,h) x 16-query-row block, d=64.
// Online softmax over 32-key blocks; P re-shaped D->A layout via LDS.
// ---------------------------------------------------------------------------
__global__ __launch_bounds__(256)
void k_attn(const bf16* __restrict__ qk, const bf16* __restrict__ vt,
            const int* __restrict__ mask, bf16* __restrict__ ao) {
  __shared__ __align__(16) bf16 plds[8][16 * 32];   // 1 KB per wave
  int warp = threadIdx.x >> 5;
  int wid = blockIdx.x * (blockDim.x >> 5) + warp;
  int bh = wid >> 7;            // 0..63
  int qb = wid & 127;           // 0..127
  int b = bh >> 4, h = bh & 15;
  int q0 = qb * 16;
  int lane = threadIdx.x & 31, hi = lane >> 4, ln = lane & 15;

  const bf16* qbase = qk + (size_t)(b * 2048 + q0) * 2048 + h * 64;     // Q rows
  const bf16* kbase = qk + (size_t)(b * 2048) * 2048 + 1024 + h * 64;   // K rows
  const bf16* vbase = vt + (size_t)(b * 16 + h) * 64 * 2048;            // [d][n]
  const int*  mbase = mask + (size_t)(b * 2048 + q0 + 8 * hi) * 2048 + ln;

  v16bf qa0 = load_a_rowmajor(qbase, 2048);        // Q, K-dim 0..31
  v16bf qa1 = load_a_rowmajor(qbase + 32, 2048);   // Q, K-dim 32..63

  v8f o0 = {}, o1 = {}, o2 = {}, o3 = {};          // O accum, 16x64
  float mrow[8], lrow[8];
#pragma unroll
  for (int r = 0; r < 8; ++r) { mrow[r] = -1e30f; lrow[r] = 0.0f; }

  for (int kb = 0; kb < 2048; kb += 32) {
    __builtin_prefetch(kbase + (size_t)(kb + 32) * 2048, 0, 1);

    // S = Q * K^T   (two 16x16 f32 tiles covering 32 keys)
    v8f s0 = {}, s1 = {};
    s0 = wmma_bf16(qa0, load_b_kmajor(kbase + (size_t)kb * 2048, 2048), s0);
    s0 = wmma_bf16(qa1, load_b_kmajor(kbase + (size_t)kb * 2048 + 32, 2048), s0);
    s1 = wmma_bf16(qa0, load_b_kmajor(kbase + (size_t)(kb + 16) * 2048, 2048), s1);
    s1 = wmma_bf16(qa1, load_b_kmajor(kbase + (size_t)(kb + 16) * 2048 + 32, 2048), s1);

    // mask fill (-1e10 where mask==1); Q was pre-scaled, no extra mul needed
#pragma unroll
    for (int r = 0; r < 8; ++r) {
      const int* mp = mbase + (size_t)r * 2048 + kb;
      if (mp[0]  == 1) s0[r] = -1e10f;
      if (mp[16] == 1) s1[r] = -1e10f;
    }

    // online softmax: row stats live in lanes sharing hi-group (width-16 xor)
#pragma unroll
    for (int r = 0; r < 8; ++r) {
      float mx = fmaxf(s0[r], s1[r]);
#pragma unroll
      for (int off = 8; off >= 1; off >>= 1)
        mx = fmaxf(mx, __shfl_xor(mx, off, 32));
      float mn   = fmaxf(mrow[r], mx);
      float corr = __expf(mrow[r] - mn);
      mrow[r] = mn;
      float e0 = __expf(s0[r] - mn);
      float e1 = __expf(s1[r] - mn);
      s0[r] = e0; s1[r] = e1;
      float rs = e0 + e1;
#pragma unroll
      for (int off = 8; off >= 1; off >>= 1)
        rs += __shfl_xor(rs, off, 32);
      lrow[r] = lrow[r] * corr + rs;
      o0[r] *= corr; o1[r] *= corr; o2[r] *= corr; o3[r] *= corr;
    }

    // P: D-layout -> A-layout via LDS (16x32 bf16 per wave)
    __syncthreads();
#pragma unroll
    for (int r = 0; r < 8; ++r) {
      plds[warp][(r + 8 * hi) * 32 + ln]      = (bf16)s0[r];
      plds[warp][(r + 8 * hi) * 32 + 16 + ln] = (bf16)s1[r];
    }
    __syncthreads();
    v16bf pa = load_a_rowmajor(plds[warp], 32);

    // O += P * V   (V^T is key-contiguous -> dword B-fragment loads)
    o0 = wmma_bf16(pa, load_b_kmajor(vbase + kb, 2048), o0);
    o1 = wmma_bf16(pa, load_b_kmajor(vbase + (size_t)16 * 2048 + kb, 2048), o1);
    o2 = wmma_bf16(pa, load_b_kmajor(vbase + (size_t)32 * 2048 + kb, 2048), o2);
    o3 = wmma_bf16(pa, load_b_kmajor(vbase + (size_t)48 * 2048 + kb, 2048), o3);
  }

  // normalize and emit head output, token-major [b*n][h*64+d] bf16
  bf16* obase = ao + (size_t)(b * 2048 + q0 + 8 * hi) * 1024 + h * 64 + ln;
#pragma unroll
  for (int r = 0; r < 8; ++r) {
    float inv = 1.0f / lrow[r];
    bf16* op = obase + (size_t)r * 1024;
    op[0]  = (bf16)(o0[r] * inv);
    op[16] = (bf16)(o1[r] * inv);
    op[32] = (bf16)(o2[r] * inv);
    op[48] = (bf16)(o3[r] * inv);
  }
}

// ---------------------------------------------------------------------------
// Output projection: out[8192,1024] = ao @ W_out + b_out  (f32 result)
// ---------------------------------------------------------------------------
__global__ __launch_bounds__(256)
void k_gemm_out(const bf16* __restrict__ ao, const bf16* __restrict__ wT,
                const float* __restrict__ bias, float* __restrict__ out) {
  const int K = 1024, tilesN = 16;  // 1024/64
  int wid = blockIdx.x * (blockDim.x >> 5) + (threadIdx.x >> 5);
  int tm = wid / tilesN, tn = wid % tilesN;
  int m0 = tm * 32, n0 = tn * 64;
  int lane = threadIdx.x & 31, hi = lane >> 4, ln = lane & 15;
  v8f acc[2][4] = {};
  for (int k0 = 0; k0 < K; k0 += 32) {
    v16bf a0 = load_a_rowmajor(ao + (size_t)m0 * K + k0, K);
    v16bf a1 = load_a_rowmajor(ao + (size_t)(m0 + 16) * K + k0, K);
#pragma unroll
    for (int j = 0; j < 4; ++j) {
      v16bf b = load_b_kmajor(wT + (size_t)(n0 + 16 * j) * K + k0, K);
      acc[0][j] = wmma_bf16(a0, b, acc[0][j]);
      acc[1][j] = wmma_bf16(a1, b, acc[1][j]);
    }
  }
#pragma unroll
  for (int j = 0; j < 4; ++j) {
    int col = n0 + 16 * j + ln;
    float bi = bias[col];
#pragma unroll
    for (int mi = 0; mi < 2; ++mi)
#pragma unroll
      for (int r = 0; r < 8; ++r)
        out[(size_t)(m0 + mi * 16 + r + 8 * hi) * 1024 + col] = acc[mi][j][r] + bi;
  }
}

// ---------------------------------------------------------------------------
// Host-side launch.  Workspace layout (88 MiB total):
//   [ 0,16) MiB  xb     bf16 x            [8192][1024]
//   [16,22) MiB  wqkvT  bf16 W_qkv^T      [3072][1024]
//   [22,24) MiB  woutT  bf16 W_out^T      [1024][1024]
//   [24,56) MiB  qk     bf16 Q|K          [8192][2048]  (Q pre-scaled)
//   [56,72) MiB  vt     bf16 V^T          [64][64][2048]
//   [72,88) MiB  ao     bf16 head output  [8192][1024]
// ---------------------------------------------------------------------------
extern "C" void kernel_launch(void* const* d_in, const int* in_sizes, int n_in,
                              void* d_out, int out_size, void* d_ws, size_t ws_size,
                              hipStream_t stream) {
  const float* x    = (const float*)d_in[0];
  const int*   mask = (const int*)d_in[1];
  const float* Wqkv = (const float*)d_in[2];
  const float* Wout = (const float*)d_in[3];
  const float* bout = (const float*)d_in[4];
  float* out = (float*)d_out;

  char* ws = (char*)d_ws;
  bf16* xb    = (bf16*)(ws);
  bf16* wqkvT = (bf16*)(ws + (16ull << 20));
  bf16* woutT = (bf16*)(ws + (22ull << 20));
  bf16* qk    = (bf16*)(ws + (24ull << 20));
  bf16* vt    = (bf16*)(ws + (56ull << 20));
  bf16* ao    = (bf16*)(ws + (72ull << 20));

  k_cvt<<<(8388608 + 255) / 256, 256, 0, stream>>>(x, xb, 8388608);
  k_cvt_t<<<(3145728 + 255) / 256, 256, 0, stream>>>(Wqkv, wqkvT, 1024, 3072);
  k_cvt_t<<<(1048576 + 255) / 256, 256, 0, stream>>>(Wout, woutT, 1024, 1024);

  // 256x48 wave tiles, 8 waves (wave32) per 256-thread block
  k_gemm_qkv<<<(256 * 48) / 8, 256, 0, stream>>>(xb, wqkvT, qk, vt);
  // 64 (b,h) x 128 query blocks = 8192 wave tiles
  k_attn<<<(64 * 128) / 8, 256, 0, stream>>>(qk, vt, mask, ao);
  // 256x16 wave tiles
  k_gemm_out<<<(256 * 16) / 8, 256, 0, stream>>>(ao, woutT, bout, out);
}
